// LengthRegulator_103079215822
// MI455X (gfx1250) — compile-verified, matrix operand order checked by reading, not run on hardware
//
#include <hip/hip_runtime.h>
#include <hip/hip_bf16.h>

// ---------------------------------------------------------------------------
// FastSpeech LengthRegulator forward for MI455X (gfx1250, wave32, WMMA).
//   B=32, T=512, C=F=384, M=2048, K=3 convs -> WMMA f32_16x16x32_f16 GEMMs.
//   Block = 64 T-rows x 384 channels; 8 waves x (3 n-tiles x 4 m-tiles).
// ---------------------------------------------------------------------------

typedef __attribute__((ext_vector_type(16))) _Float16     v16h;
typedef __attribute__((ext_vector_type(8)))  float        v8f;
typedef __attribute__((ext_vector_type(4)))  unsigned int u32x4;
typedef __attribute__((ext_vector_type(4)))  float        f32x4;

#define B_   32
#define T_   512
#define C_   384            // Cin == Cout == 384 for both conv layers
#define M_   2048
#define NCHUNK 12           // 384 / 32 K-chunks per conv tap
#define KCH  (3 * NCHUNK)   // 36 total K-chunks (3 taps)
#define WFRAG_HALVES (KCH * 24 * 32 * 16)   // 442368 halves per layer
#define MT_  64             // T-rows per block
#define LDSX_STRIDE 392     // 384 + 8 halves pad (row stride 784B -> 4-bank skew)
#define LDSO_STRIDE 388     // 384 + 4 floats pad
#define BTILE_H 512         // halves per (chunk,ntile,lane-set) fragment block: 32*16
#define BCH_H  (24 * BTILE_H) // halves per K-chunk across all 24 n-tiles

union Frag16 { v16h v; u32x4 q[2]; };

// ---------------------------------------------------------------------------
// Prep: enc * mask -> f16 activations
// ---------------------------------------------------------------------------
__global__ void prep_enc_kernel(const float* __restrict__ enc,
                                const float* __restrict__ mask,
                                _Float16* __restrict__ ench) {
  int id = blockIdx.x * blockDim.x + threadIdx.x;       // over B*T*C/4
  if (id >= B_ * T_ * C_ / 4) return;
  f32x4 e = ((const f32x4*)enc)[id];
  float mk = mask[(id * 4) / C_];                       // mask is [B,T,1]
  union { _Float16 h[4]; unsigned long long u; } p;
  p.h[0] = (_Float16)(e.x * mk);
  p.h[1] = (_Float16)(e.y * mk);
  p.h[2] = (_Float16)(e.z * mk);
  p.h[3] = (_Float16)(e.w * mk);
  ((unsigned long long*)ench)[id] = p.u;
}

// ---------------------------------------------------------------------------
// Prep: torch-layout conv weights [F, Cin, 3] -> B-fragment-linear f16 so each
// lane's 16-half WMMA B fragment is one contiguous 32-byte global read.
// Linear layout: (((tap*12+chunk)*24 + ntile)*32 + lane)*16 + j
// ---------------------------------------------------------------------------
__global__ void prep_w_kernel(const float* __restrict__ w,
                              _Float16* __restrict__ wf) {
  int id = blockIdx.x * blockDim.x + threadIdx.x;
  if (id >= WFRAG_HALVES) return;
  int j     = id & 15;
  int lane  = (id >> 4) & 31;
  int ntile = (id >> 9) % 24;
  int tc    = id / (16 * 32 * 24);      // tap*12 + chunk, 0..35
  int tap   = tc / NCHUNK;
  int chunk = tc % NCHUNK;
  // ISA 16-bit fragment element -> K mapping (mirrors A layout, N per lane)
  int v = j >> 1, h = j & 1;
  int khi = (lane >> 4) << 3;
  int K = (v < 4) ? (khi + 2 * v + h) : (16 + khi + 2 * (v - 4) + h);
  int c = chunk * 32 + K;
  int f = ntile * 16 + (lane & 15);
  wf[id] = (_Float16)w[(f * C_ + c) * 3 + tap];   // torch [F, Cin, 3]
}

// ---------------------------------------------------------------------------
// Conv1d(K=3,pad=1) + bias + ReLU + LayerNorm, fused. One block = 64 T-rows x
// all 384 channels of one batch. Each wave: 3 n-tiles x 4 m-tiles so each B
// fragment loaded from L2 is reused by 4 WMMAs. LayerNorm epilogue runs in 4
// phases of 16 rows through an LDS f32 tile that ALIASES the activation
// staging buffer (safe: first epilogue barrier orders all ldsX reads before
// the overwrite). FINAL layer fuses the 384->1 linear via wave32 reduction.
// ---------------------------------------------------------------------------
template <bool FINAL>
__global__ __launch_bounds__(256) void conv_ln_kernel(
    const _Float16* __restrict__ xin, const _Float16* __restrict__ wfrag,
    const float* __restrict__ bias, const float* __restrict__ gamma,
    const float* __restrict__ betav, _Float16* __restrict__ xout,
    const float* __restrict__ wl, const float* __restrict__ bl,
    const float* __restrict__ mask, float* __restrict__ dpo) {
  __shared__ char smem[(MT_ + 2) * LDSX_STRIDE * 2];   // 51.7 KB, aliased
  _Float16* ldsX = (_Float16*)smem;                    // (MT_+2) x 392 halves
  float*    ldsO = (float*)smem;                       // 16 x 388 f32 (reused)

  const int tid  = threadIdx.x;
  const int lane = tid & 31;
  const int wav  = tid >> 5;
  const int t0   = blockIdx.x * MT_;
  const int b    = blockIdx.y;

  // Stage rows t0-1 .. t0+MT_ (66 rows x 384 halves) into LDS, zero at edges.
  for (int i = tid; i < (MT_ + 2) * 48; i += 256) {
    int r  = i / 48;
    int cc = (i % 48) * 8;
    int gr = t0 - 1 + r;
    u32x4 val = {0u, 0u, 0u, 0u};
    if (gr >= 0 && gr < T_)
      val = *(const u32x4*)&xin[((size_t)b * T_ + gr) * C_ + cc];
    *(u32x4*)&ldsX[r * LDSX_STRIDE + cc] = val;
  }
  __syncthreads();

  const int m   = lane & 15;
  const int khi = (lane >> 4) << 3;
  const int n0  = wav * 48;

  v8f z = {0.f, 0.f, 0.f, 0.f, 0.f, 0.f, 0.f, 0.f};
  v8f acc[4][3] = {{z, z, z}, {z, z, z}, {z, z, z}, {z, z, z}};

  // Per-wave loop-invariant B base (lane's 32B fragment slice, n-tile group).
  const _Float16* bwave = wfrag + ((size_t)(wav * 3 * 32 + lane) << 4);

  // 3 shifted GEMMs (conv taps), K = 384 each, in chunks of 32.
  for (int dt = 0; dt < 3; ++dt) {
    // Loop-invariant LDS A bases for the 4 m-tiles of this tap.
    const _Float16* aptr0 = &ldsX[(0 * 16 + m + dt) * LDSX_STRIDE + khi];
    const _Float16* aptr1 = &ldsX[(1 * 16 + m + dt) * LDSX_STRIDE + khi];
    const _Float16* aptr2 = &ldsX[(2 * 16 + m + dt) * LDSX_STRIDE + khi];
    const _Float16* aptr3 = &ldsX[(3 * 16 + m + dt) * LDSX_STRIDE + khi];
    const _Float16* bptr  = bwave + (size_t)dt * NCHUNK * BCH_H;
#pragma unroll 2
    for (int ch = 0; ch < NCHUNK; ++ch, bptr += BCH_H) {
      // Load the 3 B fragments (constant immediate offsets off bptr).
      Frag16 bf[3];
#pragma unroll
      for (int tile = 0; tile < 3; ++tile) {
        const u32x4* bp = (const u32x4*)(bptr + tile * BTILE_H);
        bf[tile].q[0] = bp[0];
        bf[tile].q[1] = bp[1];
      }
      // 4 m-tiles reuse each B fragment.
#pragma unroll
      for (int mt = 0; mt < 4; ++mt) {
        const _Float16* ap =
            (mt == 0 ? aptr0 : mt == 1 ? aptr1 : mt == 2 ? aptr2 : aptr3) +
            ch * 32;
        Frag16 a;
        a.q[0] = *(const u32x4*)(ap);        // K = kc+khi .. +7
        a.q[1] = *(const u32x4*)(ap + 16);   // K = kc+16+khi .. +7
#pragma unroll
        for (int tile = 0; tile < 3; ++tile)
          acc[mt][tile] = __builtin_amdgcn_wmma_f32_16x16x32_f16(
              false, a.v, false, bf[tile].v, (short)0, acc[mt][tile], false,
              false);
      }
    }
  }

  // Epilogue: 4 phases of 16 rows. bias + ReLU -> LDS f32 tile -> LayerNorm.
  for (int mt = 0; mt < 4; ++mt) {
    __syncthreads();   // phase 0: orders all ldsX reads before aliasing write
#pragma unroll
    for (int tile = 0; tile < 3; ++tile) {
      int n = n0 + tile * 16 + m;
      float bb = bias[n];
#pragma unroll
      for (int v = 0; v < 8; ++v) {
        int row = v + ((lane >> 4) << 3);      // ISA C/D layout
        ldsO[row * LDSO_STRIDE + n] = fmaxf(acc[mt][tile][v] + bb, 0.f);
      }
    }
    __syncthreads();

    // LayerNorm over 384 channels: each wave owns 2 rows; 12 elems per lane.
#pragma unroll
    for (int rr = 0; rr < 2; ++rr) {
      int row = wav * 2 + rr;
      int t = t0 + mt * 16 + row;
      float vals[12];
      float s = 0.f, s2 = 0.f;
#pragma unroll
      for (int i = 0; i < 12; ++i) {
        float v = ldsO[row * LDSO_STRIDE + lane + 32 * i];
        vals[i] = v;
        s += v;
        s2 += v * v;
      }
#pragma unroll
      for (int off = 16; off > 0; off >>= 1) {
        s  += __shfl_xor(s, off, 32);
        s2 += __shfl_xor(s2, off, 32);
      }
      float mu   = s * (1.f / C_);
      float var  = s2 * (1.f / C_) - mu * mu;
      float rstd = rsqrtf(var + 1e-5f);
      float dot  = 0.f;
#pragma unroll
      for (int i = 0; i < 12; ++i) {
        int f = lane + 32 * i;
        float y = (vals[i] - mu) * rstd * gamma[f] + betav[f];
        if (!FINAL)
          xout[((size_t)b * T_ + t) * C_ + f] = (_Float16)y;
        else
          dot += y * wl[f];
      }
      if (FINAL) {
#pragma unroll
        for (int off = 16; off > 0; off >>= 1) dot += __shfl_xor(dot, off, 32);
        if (lane == 0)
          dpo[(size_t)b * T_ + t] = (dot + bl[0]) * mask[(size_t)b * T_ + t];
      }
    }
  }
}

// ---------------------------------------------------------------------------
// Inclusive cumsum of target durations per batch (Hillis-Steele in LDS).
// ---------------------------------------------------------------------------
__global__ void cumsum_kernel(const int* __restrict__ tgt, int* __restrict__ cs) {
  __shared__ int buf[2][T_];
  int b = blockIdx.x, tid = threadIdx.x;   // blockDim == T_ == 512
  buf[0][tid] = tgt[b * T_ + tid];
  int cur = 0;
  for (int off = 1; off < T_; off <<= 1) {
    __syncthreads();
    int v = buf[cur][tid];
    if (tid >= off) v += buf[cur][tid - off];
    buf[cur ^ 1][tid] = v;
    cur ^= 1;
  }
  __syncthreads();
  cs[b * T_ + tid] = buf[cur][tid];
}

// ---------------------------------------------------------------------------
// Expand: one wave per output frame. Binary search (searchsorted right) then
// copy 384 f32 = 3 x float4 per lane. Invalid frames write zeros.
// ---------------------------------------------------------------------------
__global__ __launch_bounds__(256) void expand_kernel(
    const float* __restrict__ enc, const int* __restrict__ cs,
    float* __restrict__ out, float* __restrict__ dec) {
  int lane = threadIdx.x & 31, wav = threadIdx.x >> 5;
  int gid = blockIdx.x * 8 + wav;
  int b = gid / M_, pos = gid % M_;
  const int* csb = cs + b * T_;
  int total = csb[T_ - 1];
  int lo = 0, hi = T_;
  while (lo < hi) {
    int mid = (lo + hi) >> 1;
    if (csb[mid] <= pos) lo = mid + 1; else hi = mid;
  }
  int idx = lo < T_ - 1 ? lo : T_ - 1;
  bool valid = pos < total;
  const f32x4* src = (const f32x4*)&enc[((size_t)b * T_ + idx) * C_];
  f32x4* dst = (f32x4*)&out[((size_t)b * M_ + pos) * C_];
  f32x4 zero = {0.f, 0.f, 0.f, 0.f};
#pragma unroll
  for (int i = 0; i < 3; ++i)
    dst[lane + 32 * i] = valid ? src[lane + 32 * i] : zero;
  if (lane == 0)
    dec[(size_t)b * M_ + pos] = valid ? (float)(pos + 1) : 0.f;
}

// ---------------------------------------------------------------------------
extern "C" void kernel_launch(void* const* d_in, const int* in_sizes, int n_in,
                              void* d_out, int out_size, void* d_ws, size_t ws_size,
                              hipStream_t stream) {
  const float* enc  = (const float*)d_in[0];
  const float* mask = (const float*)d_in[1];
  const int*   tgt  = (const int*)d_in[2];
  const float* w1   = (const float*)d_in[4];
  const float* b1   = (const float*)d_in[5];
  const float* g1   = (const float*)d_in[6];
  const float* be1  = (const float*)d_in[7];
  const float* w2   = (const float*)d_in[8];
  const float* b2   = (const float*)d_in[9];
  const float* g2   = (const float*)d_in[10];
  const float* be2  = (const float*)d_in[11];
  const float* wl   = (const float*)d_in[12];
  const float* bl   = (const float*)d_in[13];

  // Workspace layout (all chunks 256B-aligned by construction).
  char* ws = (char*)d_ws;
  size_t o = 0;
  _Float16* ench = (_Float16*)(ws + o); o += (size_t)B_ * T_ * C_ * 2;  // 12.6MB
  _Float16* w1f  = (_Float16*)(ws + o); o += (size_t)WFRAG_HALVES * 2;  // 884KB
  _Float16* w2f  = (_Float16*)(ws + o); o += (size_t)WFRAG_HALVES * 2;  // 884KB
  _Float16* x1h  = (_Float16*)(ws + o); o += (size_t)B_ * T_ * C_ * 2;  // 12.6MB
  int*      cs   = (int*)(ws + o);      o += (size_t)B_ * T_ * 4;       // 64KB

  // Output layout: [B*M*C output][B*M dec_pos][B*T dpo], all f32.
  float* out = (float*)d_out;
  float* dec = out + (size_t)B_ * M_ * C_;
  float* dpo = dec + (size_t)B_ * M_;

  prep_enc_kernel<<<(B_ * T_ * C_ / 4 + 255) / 256, 256, 0, stream>>>(enc, mask, ench);
  prep_w_kernel<<<(WFRAG_HALVES + 255) / 256, 256, 0, stream>>>(w1, w1f);
  prep_w_kernel<<<(WFRAG_HALVES + 255) / 256, 256, 0, stream>>>(w2, w2f);

  conv_ln_kernel<false><<<dim3(T_ / MT_, B_), 256, 0, stream>>>(
      ench, w1f, b1, g1, be1, x1h, nullptr, nullptr, nullptr, nullptr);
  conv_ln_kernel<true><<<dim3(T_ / MT_, B_), 256, 0, stream>>>(
      x1h, w2f, b2, g2, be2, nullptr, wl, bl, mask, dpo);

  cumsum_kernel<<<B_, T_, 0, stream>>>(tgt, cs);
  expand_kernel<<<B_ * M_ / 8, 256, 0, stream>>>(enc, cs, out, dec);
}
